// RNNModule_83726092468348
// MI455X (gfx1250) — compile-verified, matrix operand order checked by reading, not run on hardware
//
#include <hip/hip_runtime.h>
#include <hip/hip_bf16.h>

// Problem dims (hardcoded from reference)
#define B_DIM   2048
#define Z_DIM   512
#define H_DIM   384
#define G_DIM   1536   // 4*H
#define T_FR    32
#define KCAT    896    // Z + H
#define HC_DIM  768    // 2*H
#define W_RES   0.2f

typedef __attribute__((ext_vector_type(16))) __bf16        v16bf;
typedef __attribute__((ext_vector_type(8)))  float         v8f;
typedef __attribute__((ext_vector_type(8)))  unsigned int  v8u;
typedef int v4i __attribute__((vector_size(16)));   // matches builtin's V4i

#if __has_builtin(__builtin_amdgcn_global_load_async_to_lds_b128) && \
    __has_builtin(__builtin_amdgcn_s_wait_asynccnt)
#define USE_ASYNC_LDS 1
#else
#define USE_ASYNC_LDS 0
#endif

__device__ __forceinline__ unsigned short f2bf(float f) {
  unsigned int u = __float_as_uint(f);
  u += 0x7FFFu + ((u >> 16) & 1u);   // round-to-nearest-even
  return (unsigned short)(u >> 16);
}
__device__ __forceinline__ float sigf(float x) { return 1.0f / (1.0f + expf(-x)); }

// ---------------------------------------------------------------------------
// bf16 WMMA GEMM:  C[M,N] = A[M,K] * B[N,K]^T (+bias) (optional ReLU)
// A,B row-major bf16 (K contiguous). Output fp32 (C) or bf16 (Cbf) w/ stride ldc.
// Block tile 128x64, K-step 64, 256 threads = 8 waves.
// Double-buffered LDS; async global->LDS copies (ASYNCcnt) when available.
// One barrier per K-step: wait_async -> barrier -> prefetch(other) -> compute(cur).
// Global addresses are per-thread pointers stepped by 128B/iter (no k0 rescale).
// Requires: M%128==0, N%64==0, K%64==0 (true for all call sites).
// lda2/ldb2 are row strides in DWORDS (bf16 pairs).
// ---------------------------------------------------------------------------
__global__ __launch_bounds__(256)
void gemm_bf16_wmma(const unsigned short* __restrict__ A, int lda2,
                    const unsigned short* __restrict__ B, int ldb2,
                    float* __restrict__ C, unsigned short* __restrict__ Cbf,
                    int ldc, const float* __restrict__ bias, int K, int relu)
{
  // 64 bf16 = 32 dwords per row; pad stride to 36 dwords (16B-aligned, bank spread)
  __shared__ unsigned int ldsA[2][128 * 36];
  __shared__ unsigned int ldsB[2][64 * 36];

  const int tid  = threadIdx.x;
  const int wave = tid >> 5;
  const int lane = tid & 31;
  const int bm   = blockIdx.y * 128;
  const int bn   = blockIdx.x * 64;

  const unsigned int* Ad = (const unsigned int*)A;
  const unsigned int* Bd = (const unsigned int*)B;

  v8f acc[4];
#pragma unroll
  for (int t = 0; t < 4; ++t)
#pragma unroll
    for (int j = 0; j < 8; ++j) acc[t][j] = 0.0f;

  const int mrow = wave * 16 + (lane & 15);
  const int dsel = (lane >> 4) * 4;   // K-half select per WMMA 16-bit A/B layout

  // Per-thread staging pointers (stepped by 32 dwords per K-step) + LDS offsets
  const unsigned int* aP[4]; int offA[4];
  const unsigned int* bP[2]; int offB[2];
#pragma unroll
  for (int i = 0; i < 4; ++i) {               // A: 128x64 bf16 = 4096 dwords
    int d = (tid + i * 256) * 4;
    int r = d >> 5, c = d & 31;
    aP[i]   = Ad + (size_t)(bm + r) * lda2 + c;
    offA[i] = r * 36 + c;
  }
#pragma unroll
  for (int i = 0; i < 2; ++i) {               // B: 64x64 bf16 = 2048 dwords
    int d = (tid + i * 256) * 4;
    int r = d >> 5, c = d & 31;
    bP[i]   = Bd + (size_t)(bn + r) * ldb2 + c;
    offB[i] = r * 36 + c;
  }

  auto stage_tile = [&](int buf) {
#if USE_ASYNC_LDS
#pragma unroll
    for (int i = 0; i < 4; ++i) {
      __builtin_amdgcn_global_load_async_to_lds_b128(
          (__attribute__((address_space(1))) v4i*)aP[i],
          (__attribute__((address_space(3))) v4i*)&ldsA[buf][offA[i]], 0, 0);
      aP[i] += 32;
    }
#pragma unroll
    for (int i = 0; i < 2; ++i) {
      __builtin_amdgcn_global_load_async_to_lds_b128(
          (__attribute__((address_space(1))) v4i*)bP[i],
          (__attribute__((address_space(3))) v4i*)&ldsB[buf][offB[i]], 0, 0);
      bP[i] += 32;
    }
#else
    uint4 ta[4], tb[2];
#pragma unroll
    for (int i = 0; i < 4; ++i) { ta[i] = *(const uint4*)aP[i]; aP[i] += 32; }
#pragma unroll
    for (int i = 0; i < 2; ++i) { tb[i] = *(const uint4*)bP[i]; bP[i] += 32; }
#pragma unroll
    for (int i = 0; i < 4; ++i) *(uint4*)&ldsA[buf][offA[i]] = ta[i];
#pragma unroll
    for (int i = 0; i < 2; ++i) *(uint4*)&ldsB[buf][offB[i]] = tb[i];
#endif
  };

  const int nk = K >> 6;
  stage_tile(0);

  for (int it = 0; it < nk; ++it) {
#if USE_ASYNC_LDS
    __builtin_amdgcn_s_wait_asynccnt(0);  // my async writes for buf[cur] landed in LDS
#endif
    __syncthreads();                      // all waves' tile visible; prev readers done
    if (it + 1 < nk) stage_tile((it + 1) & 1);

    const unsigned int* bufA = ldsA[it & 1];
    const unsigned int* bufB = ldsB[it & 1];
#pragma unroll
    for (int ks = 0; ks < 2; ++ks) {      // two K=32 sub-steps
      const unsigned int* ar = bufA + mrow * 36 + ks * 16;
      uint4 a0 = *(const uint4*)(ar + dsel);
      uint4 a1 = *(const uint4*)(ar + 8 + dsel);
      uint4 b0[4], b1[4];
#pragma unroll
      for (int t = 0; t < 4; ++t) {
        const unsigned int* br = bufB + (t * 16 + (lane & 15)) * 36 + ks * 16;
        b0[t] = *(const uint4*)(br + dsel);
        b1[t] = *(const uint4*)(br + 8 + dsel);
      }
      v8u au;
      au[0] = a0.x; au[1] = a0.y; au[2] = a0.z; au[3] = a0.w;
      au[4] = a1.x; au[5] = a1.y; au[6] = a1.z; au[7] = a1.w;
      v16bf av = __builtin_bit_cast(v16bf, au);
#pragma unroll
      for (int t = 0; t < 4; ++t) {
        v8u bu;
        bu[0] = b0[t].x; bu[1] = b0[t].y; bu[2] = b0[t].z; bu[3] = b0[t].w;
        bu[4] = b1[t].x; bu[5] = b1[t].y; bu[6] = b1[t].z; bu[7] = b1[t].w;
        v16bf bv = __builtin_bit_cast(v16bf, bu);
        acc[t] = __builtin_amdgcn_wmma_f32_16x16x32_bf16(
            false, av, false, bv, (short)0, acc[t], false, false);
      }
    }
  }

  // Epilogue. D layout: VGPR r -> M = r (+8 for lanes 16..31), N = lane&15.
  const int ncol0 = bn + (lane & 15);
  const int rbase = bm + wave * 16 + (lane >> 4) * 8;
#pragma unroll
  for (int t = 0; t < 4; ++t) {
    int col = ncol0 + t * 16;
    float bb = bias ? bias[col] : 0.0f;
#pragma unroll
    for (int r = 0; r < 8; ++r) {
      float v = acc[t][r] + bb;
      if (relu) v = fmaxf(v, 0.0f);
      if (Cbf) Cbf[(size_t)(rbase + r) * ldc + col] = f2bf(v);
      else     C  [(size_t)(rbase + r) * ldc + col] = v;
    }
  }
}

// ---------------------------------------------------------------------------
// Elementwise / packing kernels
// ---------------------------------------------------------------------------
__global__ void pack_bf16_kernel(const float* __restrict__ src,
                                 unsigned short* __restrict__ dst, int n) {
  int i = blockIdx.x * blockDim.x + threadIdx.x;
  if (i < n) dst[i] = f2bf(src[i]);
}

// Wcat[G_DIM x KCAT] = [Wih | Whh] rows
__global__ void pack_cat_kernel(const float* __restrict__ Wih,
                                const float* __restrict__ Whh,
                                unsigned short* __restrict__ dst) {
  int i = blockIdx.x * blockDim.x + threadIdx.x;
  if (i >= G_DIM * KCAT) return;
  int r = i / KCAT, c = i % KCAT;
  float v = (c < Z_DIM) ? Wih[(size_t)r * Z_DIM + c]
                        : Whh[(size_t)r * H_DIM + (c - Z_DIM)];
  dst[i] = f2bf(v);
}

// wT[Z_DIM x H_DIM] = w[H_DIM x Z_DIM]^T
__global__ void pack_wT_kernel(const float* __restrict__ w,
                               unsigned short* __restrict__ dst) {
  int i = blockIdx.x * blockDim.x + threadIdx.x;
  if (i >= Z_DIM * H_DIM) return;
  int n = i / H_DIM, k = i % H_DIM;
  dst[i] = f2bf(w[(size_t)k * Z_DIM + n]);
}

// out_full[0]=z, e[0]=z, A_cat[:,0:Z]=bf16(z)
__global__ void init_z_kernel(const float* __restrict__ z,
                              float* __restrict__ out, float* __restrict__ e,
                              unsigned short* __restrict__ Acat) {
  int i = blockIdx.x * blockDim.x + threadIdx.x;
  if (i >= B_DIM * Z_DIM) return;
  int b = i / Z_DIM, zc = i % Z_DIM;
  float v = z[i];
  out[(size_t)(b * T_FR) * Z_DIM + zc]     = v;
  e[(size_t)(b * (T_FR - 1)) * Z_DIM + zc] = v;
  Acat[(size_t)b * KCAT + zc]              = f2bf(v);
}

// LSTM cell nonlinearity. gates row order: i, f, g, o (each H_DIM wide).
__global__ void lstm_elem_kernel(const float* __restrict__ gates,
                                 const float* __restrict__ bih,
                                 const float* __restrict__ bhh,
                                 float* __restrict__ cbuf,
                                 unsigned short* __restrict__ Acat,
                                 unsigned short* __restrict__ hc, int first) {
  int i = blockIdx.x * blockDim.x + threadIdx.x;
  if (i >= B_DIM * H_DIM) return;
  int b = i / H_DIM, j = i % H_DIM;
  const float* g = gates + (size_t)b * G_DIM;
  float gi = g[j]             + bih[j]             + bhh[j];
  float gf = g[H_DIM + j]     + bih[H_DIM + j]     + bhh[H_DIM + j];
  float gg = g[2 * H_DIM + j] + bih[2 * H_DIM + j] + bhh[2 * H_DIM + j];
  float go = g[3 * H_DIM + j] + bih[3 * H_DIM + j] + bhh[3 * H_DIM + j];
  float cold = first ? 0.0f : cbuf[i];
  float cnew = sigf(gf) * cold + sigf(gi) * tanhf(gg);
  float hnew = sigf(go) * tanhf(cnew);
  cbuf[i] = cnew;
  Acat[(size_t)b * KCAT + Z_DIM + j] = f2bf(hnew);
  hc[(size_t)b * HC_DIM + j]         = f2bf(hnew);
  hc[(size_t)b * HC_DIM + H_DIM + j] = f2bf(cnew);
}

// out[t+1] = out[t] + 0.2*tanh(mulpre); also writes e[t+1] and bf16 A_cat[:,0:Z]
__global__ void residual_elem_kernel(const float* __restrict__ mulpre,
                                     float* __restrict__ out, float* __restrict__ e,
                                     unsigned short* __restrict__ Acat, int t) {
  int i = blockIdx.x * blockDim.x + threadIdx.x;
  if (i >= B_DIM * Z_DIM) return;
  int b = i / Z_DIM, zc = i % Z_DIM;
  float m    = tanhf(mulpre[i]);
  float prev = out[(size_t)(b * T_FR + t) * Z_DIM + zc];
  float on   = prev + W_RES * m;
  out[(size_t)(b * T_FR + t + 1) * Z_DIM + zc] = on;
  if (t + 1 < T_FR - 1)
    e[(size_t)(b * (T_FR - 1) + t + 1) * Z_DIM + zc] = on;
  Acat[(size_t)b * KCAT + zc] = f2bf(on);
}

// ---------------------------------------------------------------------------
extern "C" void kernel_launch(void* const* d_in, const int* in_sizes, int n_in,
                              void* d_out, int out_size, void* d_ws, size_t ws_size,
                              hipStream_t stream) {
  (void)in_sizes; (void)n_in; (void)out_size; (void)ws_size;
  const float* z       = (const float*)d_in[0];
  const float* Wih_enc = (const float*)d_in[1];
  const float* Whh_enc = (const float*)d_in[2];   (void)Whh_enc; // h0=0 -> only bias matters
  const float* bih_enc = (const float*)d_in[3];
  const float* bhh_enc = (const float*)d_in[4];
  const float* Wih     = (const float*)d_in[5];
  const float* Whh     = (const float*)d_in[6];
  const float* bih     = (const float*)d_in[7];
  const float* bhh     = (const float*)d_in[8];
  const float* w       = (const float*)d_in[9];
  const float* bvec    = (const float*)d_in[10];
  const float* fc1_w   = (const float*)d_in[11];
  const float* fc1_b   = (const float*)d_in[12];
  const float* fc2_w   = (const float*)d_in[13];
  const float* fc2_b   = (const float*)d_in[14];

  float* out  = (float*)d_out;
  float* e    = out + (size_t)B_DIM * T_FR * Z_DIM;
  float* erec = e   + (size_t)B_DIM * (T_FR - 1) * Z_DIM;

  // Workspace carve-up (all sizes are multiples of 256B)
  char* p = (char*)d_ws;
  unsigned short* Acat   = (unsigned short*)p; p += (size_t)B_DIM * KCAT * 2;
  unsigned short* hcbuf  = (unsigned short*)p; p += (size_t)B_DIM * HC_DIM * 2;
  float*          gates  = (float*)p;          p += (size_t)B_DIM * G_DIM * 4;
  float*          cbuf   = (float*)p;          p += (size_t)B_DIM * H_DIM * 4;
  float*          mulpre = (float*)p;          p += (size_t)B_DIM * Z_DIM * 4;
  unsigned short* t1bf   = (unsigned short*)p; p += (size_t)B_DIM * Z_DIM * 2;
  unsigned short* wencbf = (unsigned short*)p; p += (size_t)G_DIM * Z_DIM * 2;
  unsigned short* wcatbf = (unsigned short*)p; p += (size_t)G_DIM * KCAT * 2;
  unsigned short* wtbf   = (unsigned short*)p; p += (size_t)Z_DIM * H_DIM * 2;
  unsigned short* fc1bf  = (unsigned short*)p; p += (size_t)Z_DIM * HC_DIM * 2;
  unsigned short* fc2bf  = (unsigned short*)p; p += (size_t)Z_DIM * Z_DIM * 2;

  const int TB = 256;
  auto blocks = [](int n) { return (n + 255) / 256; };

  // Weight packing (fp32 -> bf16, once per launch)
  pack_bf16_kernel<<<blocks(G_DIM * Z_DIM), TB, 0, stream>>>(Wih_enc, wencbf, G_DIM * Z_DIM);
  pack_cat_kernel<<<blocks(G_DIM * KCAT), TB, 0, stream>>>(Wih, Whh, wcatbf);
  pack_wT_kernel<<<blocks(Z_DIM * H_DIM), TB, 0, stream>>>(w, wtbf);
  pack_bf16_kernel<<<blocks(Z_DIM * HC_DIM), TB, 0, stream>>>(fc1_w, fc1bf, Z_DIM * HC_DIM);
  pack_bf16_kernel<<<blocks(Z_DIM * Z_DIM), TB, 0, stream>>>(fc2_w, fc2bf, Z_DIM * Z_DIM);

  init_z_kernel<<<blocks(B_DIM * Z_DIM), TB, 0, stream>>>(z, out, e, Acat);

  const dim3 gridG(G_DIM / 64, B_DIM / 128);   // N=1536 GEMMs
  const dim3 gridZ(Z_DIM / 64, B_DIM / 128);   // N=512  GEMMs

  // Encoder: gates = z @ Wih_enc^T  (A = A_cat cols [0,Z), lda = KCAT, K=512)
  gemm_bf16_wmma<<<gridG, TB, 0, stream>>>(Acat, KCAT / 2, wencbf, Z_DIM / 2,
                                           gates, nullptr, G_DIM, nullptr, Z_DIM, 0);
  lstm_elem_kernel<<<blocks(B_DIM * H_DIM), TB, 0, stream>>>(
      gates, bih_enc, bhh_enc, cbuf, Acat, hcbuf, /*first=*/1);

  for (int t = 0; t < T_FR - 1; ++t) {
    // gates = [out | h] @ [Wih | Whh]^T   (K = 896)
    gemm_bf16_wmma<<<gridG, TB, 0, stream>>>(Acat, KCAT / 2, wcatbf, KCAT / 2,
                                             gates, nullptr, G_DIM, nullptr, KCAT, 0);
    lstm_elem_kernel<<<blocks(B_DIM * H_DIM), TB, 0, stream>>>(
        gates, bih, bhh, cbuf, Acat, hcbuf, 0);

    // mulpre = h2 @ w + b   (A = A_cat cols [Z, Z+H), K = 384)
    gemm_bf16_wmma<<<gridZ, TB, 0, stream>>>(Acat + Z_DIM, KCAT / 2, wtbf, H_DIM / 2,
                                             mulpre, nullptr, Z_DIM, bvec, H_DIM, 0);
    residual_elem_kernel<<<blocks(B_DIM * Z_DIM), TB, 0, stream>>>(mulpre, out, e, Acat, t);

    // t1bf = bf16(relu(hc @ fc1_w^T + fc1_b))   (K = 768, fused bf16 epilogue)
    gemm_bf16_wmma<<<gridZ, TB, 0, stream>>>(hcbuf, HC_DIM / 2, fc1bf, HC_DIM / 2,
                                             nullptr, t1bf, Z_DIM, fc1_b, HC_DIM, 1);

    // e_rec[b*31 + t] = t1 @ fc2_w^T + fc2_b  (strided output, ldc = 31*Z)
    gemm_bf16_wmma<<<gridZ, TB, 0, stream>>>(t1bf, Z_DIM / 2, fc2bf, Z_DIM / 2,
                                             erec + (size_t)t * Z_DIM, nullptr,
                                             (T_FR - 1) * Z_DIM, fc2_b, Z_DIM, 0);
  }
}
